// ProjectedAdaptiveLogSoftmax_75642964017948
// MI455X (gfx1250) — matrix operand order repaired and checked by reading, not compiled, not run on hardware
//
#include <hip/hip_runtime.h>
#include <hip/hip_bf16.h>

typedef __attribute__((ext_vector_type(16))) _Float16 v16h;
typedef __attribute__((ext_vector_type(8)))  _Float16 v8h;
typedef __attribute__((ext_vector_type(8)))  float    v8f;
typedef __attribute__((ext_vector_type(4)))  float    v4f;

#define NEG_BIG (-1e30f)
#define NTOK 511      // B*(S-1)
#define NTOK_PAD 512
#define HEADV 20000

// ---------------------------------------------------------------------------
// Operand loaders for CDNA5 WMMA 16-bit operands (wave32); A and B share contiguous k order k = 16h + s.
// A (16x32, M=lane&15): lanes 0-15 hold K = 0..15, lanes 16-31 hold
// K = 16..31  -> loaded as two contiguous 8-half runs per lane.
// B (32x16, N=lane&15): lanes 0-15 hold K = 0..15, lanes 16-31 hold K = 16..31
// -> one contiguous 16-half run per lane.
// ---------------------------------------------------------------------------
__device__ __forceinline__ v16h load_a_sh(const _Float16* sA, int stride,
                                          int lane, int k0, int d) {
  const int r    = lane & 15;
  const int base = (lane < 16) ? 0 : 16;
  const _Float16* p = sA + r * stride;
  const int kb0 = k0 + base;
  const int kb1 = k0 + 8 + base;
  v16h a;
  if (kb0 + 8 <= d) {
    v8h x = *(const v8h*)(p + kb0);
#pragma unroll
    for (int e = 0; e < 8; ++e) a[e] = x[e];
  } else {
#pragma unroll
    for (int e = 0; e < 8; ++e) a[e] = (_Float16)0.f;
  }
  if (kb1 + 8 <= d) {
    v8h x = *(const v8h*)(p + kb1);
#pragma unroll
    for (int e = 0; e < 8; ++e) a[8 + e] = x[e];
  } else {
#pragma unroll
    for (int e = 0; e < 8; ++e) a[8 + e] = (_Float16)0.f;
  }
  return a;
}

// B from a weight ROW (logits = H @ W^T): lane n reads W[row n, k..k+15] f32.
__device__ __forceinline__ v16h load_b_row(const float* wrow, int lane, int k0, int d) {
  const int kb = k0 + ((lane < 16) ? 0 : 16);
  v16h b;
  if (kb + 16 <= d) {
    v4f f0 = *(const v4f*)(wrow + kb);
    v4f f1 = *(const v4f*)(wrow + kb + 4);
    v4f f2 = *(const v4f*)(wrow + kb + 8);
    v4f f3 = *(const v4f*)(wrow + kb + 12);
#pragma unroll
    for (int e = 0; e < 4; ++e) {
      b[e]      = (_Float16)f0[e];
      b[4 + e]  = (_Float16)f1[e];
      b[8 + e]  = (_Float16)f2[e];
      b[12 + e] = (_Float16)f3[e];
    }
  } else {
#pragma unroll
    for (int e = 0; e < 16; ++e) b[e] = (_Float16)0.f;
  }
  return b;
}

// B from a matrix COLUMN (H = X @ P): lane n reads P[k, col], strided, coalesced.
__device__ __forceinline__ v16h load_b_col(const float* P, int d, int col,
                                           int lane, int k0) {
  const int kb = k0 + ((lane < 16) ? 0 : 16);
  v16h b;
#pragma unroll
  for (int e = 0; e < 16; ++e) b[e] = (_Float16)P[(size_t)(kb + e) * d + col];
  return b;
}

__device__ __forceinline__ void lse_merge(float& m, float& s, float m2, float s2) {
  float mn = fmaxf(m, m2);
  s = s * __expf(m - mn) + s2 * __expf(m2 - mn);
  m = mn;
}

// ---------------------------------------------------------------------------
// Per-token gather targets.
// ---------------------------------------------------------------------------
__global__ void setup_q(const int* __restrict__ labels, int* __restrict__ qh,
                        int* __restrict__ q1, int* __restrict__ q2,
                        int* __restrict__ q3) {
  int t = threadIdx.x;
  if (t >= NTOK_PAD) return;
  if (t >= NTOK) { qh[t] = -3; q1[t] = -3; q2[t] = -3; q3[t] = -3; return; }
  int y = labels[t + 1];
  int c = (y < 20000) ? 0 : (y < 40000) ? 1 : (y < 200000) ? 2 : 3;
  qh[t] = (c == 0) ? y : (HEADV + c - 1);
  q1[t] = (c == 1) ? (y - 20000)  : -3;
  q2[t] = (c == 2) ? (y - 40000)  : -3;
  q3[t] = (c == 3) ? (y - 200000) : -3;
}

// ---------------------------------------------------------------------------
// H_s = X @ proj_s  (all four projections, f16 output).
// grid = (32 Mtiles, 12 col-groups: 8 for d=1024, 2 for 256, 1 for 64, 1 for 16)
// ---------------------------------------------------------------------------
__global__ __launch_bounds__(256) void proj_gemm(
    const float* __restrict__ X,
    const float* __restrict__ P0, const float* __restrict__ P1,
    const float* __restrict__ P2, const float* __restrict__ P3,
    _Float16* __restrict__ H0, _Float16* __restrict__ H1,
    _Float16* __restrict__ H2, _Float16* __restrict__ H3) {
  __shared__ _Float16 sX[16 * 1032];
  const int tid   = threadIdx.x;
  const int lane  = tid & 31;
  const int wave  = tid >> 5;
  const bool hiH  = lane >= 16;
  const int mtile = blockIdx.x;
  const int yb    = blockIdx.y;

  int seg, cg;
  if (yb < 8)       { seg = 0; cg = yb; }
  else if (yb < 10) { seg = 1; cg = yb - 8; }
  else if (yb == 10){ seg = 2; cg = 0; }
  else              { seg = 3; cg = 0; }

  const float* P = (seg == 0) ? P0 : (seg == 1) ? P1 : (seg == 2) ? P2 : P3;
  _Float16*   H  = (seg == 0) ? H0 : (seg == 1) ? H1 : (seg == 2) ? H2 : H3;
  const int   d  = (seg == 0) ? 1024 : (seg == 1) ? 256 : (seg == 2) ? 64 : 16;

  for (int idx = tid; idx < 16 * 1024; idx += 256) {
    int r = idx >> 10;
    int k = idx & 1023;
    sX[r * 1032 + k] = (_Float16)X[(size_t)(mtile * 16 + r) * 1024 + k];
  }
  __syncthreads();

  const int colBase = cg * 128 + wave * 16;
  if (colBase >= d) return;  // wave-uniform exit
  const int col = colBase + (lane & 15);

  v8f c;
#pragma unroll
  for (int e = 0; e < 8; ++e) c[e] = 0.f;

  for (int k0 = 0; k0 < 1024; k0 += 32) {
    v16h a = load_a_sh(sX, 1032, lane, k0, 1024);
    v16h b = load_b_col(P, d, col, lane, k0);
    c = __builtin_amdgcn_wmma_f32_16x16x32_f16(false, a, false, b, (short)0, c,
                                               false, false);
  }

#pragma unroll
  for (int j = 0; j < 8; ++j) {
    int r = j + (hiH ? 8 : 0);
    H[(size_t)(mtile * 16 + r) * d + col] = (_Float16)c[j];
  }
}

// ---------------------------------------------------------------------------
// Logits chunk + online log-sum-exp partials + label-logit gather.
// Workgroup owns 64 tokens (4 M-subtiles resident in LDS) so each B fetch
// feeds 4 WMMAs -> 4x less weight traffic. grid = (8 Mblocks, nChunks).
// Dynamic LDS carve: sA[64*ldsStride] f16 | sPm[8][64] | sPs[8][64] | sQ[64].
// ---------------------------------------------------------------------------
__global__ __launch_bounds__(256) void lse_gemm(
    const _Float16* __restrict__ H, int d, int ldsStride,
    const float* __restrict__ W,  const float* __restrict__ Bb, int vMain,
    const float* __restrict__ Wx, const float* __restrict__ Bx, int vTotal,
    int iters,
    const int* __restrict__ qcol, float* __restrict__ lyOut,
    float* __restrict__ partM, float* __restrict__ partS) {
  extern __shared__ char smemRaw[];
  _Float16* sA = (_Float16*)smemRaw;
  float* sPm = (float*)(smemRaw + (size_t)64 * ldsStride * 2);
  float* sPs = sPm + 8 * 64;
  int*   sQ  = (int*)(sPs + 8 * 64);

  const int tid   = threadIdx.x;
  const int lane  = tid & 31;
  const int wave  = tid >> 5;
  const bool hiH  = lane >= 16;
  const int mbase = blockIdx.x * 64;
  const int chunk = blockIdx.y;
  const int chunkStart = chunk * (iters * 128);

  // Cooperative A stage: 64 rows x d halves, 16B vector units.
  const int blk = d >> 3;
  for (int idx = tid; idx < 64 * blk; idx += 256) {
    int r  = idx / blk;
    int kb = idx - r * blk;
    *(v8h*)(sA + r * ldsStride + kb * 8) =
        *(const v8h*)(H + (size_t)(mbase + r) * d + kb * 8);
  }
  if (tid < 64) {
    int tok = mbase + tid;
    sQ[tid] = (tok < NTOK) ? qcol[tok] : -3;
  }
  __syncthreads();

  float mRun[4][8], sRun[4][8];
#pragma unroll
  for (int st = 0; st < 4; ++st)
#pragma unroll
    for (int j = 0; j < 8; ++j) { mRun[st][j] = NEG_BIG; sRun[st][j] = 0.f; }

  for (int it = 0; it < iters; ++it) {
    const int col = chunkStart + it * 128 + wave * 16 + (lane & 15);
    const bool colValid = col < vTotal;
    const int colC = colValid ? col : (vTotal - 1);
    const float* wrow = (colC < vMain) ? (W + (size_t)colC * d)
                                       : (Wx + (size_t)(colC - vMain) * d);
    const float bval  = (colC < vMain) ? Bb[colC] : Bx[colC - vMain];

    if (it + 1 < iters) {  // hint next iteration's weight rows
      int ncol = col + 128;
      int ncolC = (ncol < vTotal) ? ncol : (vTotal - 1);
      const float* nrow = (ncolC < vMain) ? (W + (size_t)ncolC * d)
                                          : (Wx + (size_t)(ncolC - vMain) * d);
      __builtin_prefetch(nrow, 0, 1);
    }

    v8f c[4];
#pragma unroll
    for (int st = 0; st < 4; ++st)
#pragma unroll
      for (int e = 0; e < 8; ++e) c[st][e] = 0.f;

    for (int k0 = 0; k0 < d; k0 += 32) {
      v16h b = load_b_row(wrow, lane, k0, d);
#pragma unroll
      for (int st = 0; st < 4; ++st) {
        v16h a = load_a_sh(sA + st * 16 * ldsStride, ldsStride, lane, k0, d);
        c[st] = __builtin_amdgcn_wmma_f32_16x16x32_f16(false, a, false, b,
                                                       (short)0, c[st], false,
                                                       false);
      }
    }

#pragma unroll
    for (int st = 0; st < 4; ++st) {
#pragma unroll
      for (int j = 0; j < 8; ++j) {
        float v = colValid ? (c[st][j] + bval) : NEG_BIG;
        int tloc = st * 16 + j + (hiH ? 8 : 0);
        if (colValid && sQ[tloc] == col) lyOut[mbase + tloc] = v;
        float m16 = v;
#pragma unroll
        for (int msk = 1; msk <= 8; msk <<= 1)
          m16 = fmaxf(m16, __shfl_xor(m16, msk, 32));
        float e = __expf(v - m16);
#pragma unroll
        for (int msk = 1; msk <= 8; msk <<= 1) e += __shfl_xor(e, msk, 32);
        lse_merge(mRun[st][j], sRun[st][j], m16, e);
      }
    }
  }

  if ((lane & 15) == 0) {
#pragma unroll
    for (int st = 0; st < 4; ++st)
#pragma unroll
      for (int j = 0; j < 8; ++j) {
        int t = st * 16 + j + (hiH ? 8 : 0);
        sPm[wave * 64 + t] = mRun[st][j];
        sPs[wave * 64 + t] = sRun[st][j];
      }
  }
  __syncthreads();
  if (tid < 64) {
    float m = NEG_BIG, s = 0.f;
#pragma unroll
    for (int w = 0; w < 8; ++w) lse_merge(m, s, sPm[w * 64 + tid], sPs[w * 64 + tid]);
    partM[(size_t)chunk * NTOK_PAD + mbase + tid] = m;
    partS[(size_t)chunk * NTOK_PAD + mbase + tid] = s;
  }
}

// ---------------------------------------------------------------------------
// Final per-token NLL assembly.
// ---------------------------------------------------------------------------
__global__ void finalize_nll(
    const int* __restrict__ labels,
    const float* __restrict__ pmH, const float* __restrict__ psH, int nChH,
    const float* __restrict__ pm1, const float* __restrict__ ps1, int nCh1,
    const float* __restrict__ pm2, const float* __restrict__ ps2, int nCh2,
    const float* __restrict__ pm3, const float* __restrict__ ps3, int nCh3,
    const float* __restrict__ lyHead, const float* __restrict__ lyTail,
    float* __restrict__ out) {
  int t = blockIdx.x * blockDim.x + threadIdx.x;
  if (t >= NTOK) return;
  float m = NEG_BIG, s = 0.f;
  for (int c = 0; c < nChH; ++c)
    lse_merge(m, s, pmH[(size_t)c * NTOK_PAD + t], psH[(size_t)c * NTOK_PAD + t]);
  float logZh = m + __logf(s);
  float lph = lyHead[t] - logZh;

  int y = labels[t + 1];
  float nll;
  if (y < HEADV) {
    nll = -lph;
  } else {
    int c = (y < 40000) ? 1 : (y < 200000) ? 2 : 3;
    const float* pm = (c == 1) ? pm1 : (c == 2) ? pm2 : pm3;
    const float* ps = (c == 1) ? ps1 : (c == 2) ? ps2 : ps3;
    int nCh = (c == 1) ? nCh1 : (c == 2) ? nCh2 : nCh3;
    float m2 = NEG_BIG, s2 = 0.f;
    for (int cc = 0; cc < nCh; ++cc)
      lse_merge(m2, s2, pm[(size_t)cc * NTOK_PAD + t], ps[(size_t)cc * NTOK_PAD + t]);
    float logZt = m2 + __logf(s2);
    nll = -(lph + (lyTail[t] - logZt));
  }
  out[t] = nll;
}

// ---------------------------------------------------------------------------
extern "C" void kernel_launch(void* const* d_in, const int* in_sizes, int n_in,
                              void* d_out, int out_size, void* d_ws,
                              size_t ws_size, hipStream_t stream) {
  const float* X   = (const float*)d_in[0];   // [512,1024]
  const int*   lab = (const int*)d_in[1];     // [512]
  const float* P0  = (const float*)d_in[2];
  const float* W0  = (const float*)d_in[3];
  const float* B0  = (const float*)d_in[4];
  const float* P1  = (const float*)d_in[5];
  const float* W1  = (const float*)d_in[6];
  const float* B1  = (const float*)d_in[7];
  const float* P2  = (const float*)d_in[8];
  const float* W2  = (const float*)d_in[9];
  const float* B2  = (const float*)d_in[10];
  const float* P3  = (const float*)d_in[11];
  const float* W3  = (const float*)d_in[12];
  const float* B3  = (const float*)d_in[13];
  const float* CW  = (const float*)d_in[14];
  const float* CB  = (const float*)d_in[15];
  float* out = (float*)d_out;

  const int nChH = 20;   // ceil(20003/1024),  8 iters/chunk
  const int nCh1 = 20;   // ceil(20000/1024),  8 iters/chunk
  const int nCh2 = 79;   // ceil(160000/2048), 16 iters/chunk
  const int nCh3 = 34;   // ceil(67735/2048),  16 iters/chunk

  char* p = (char*)d_ws;
  auto alloc = [&](size_t bytes) -> void* {
    void* r = (void*)p;
    p += (bytes + 255) & ~(size_t)255;
    return r;
  };
  _Float16* H0 = (_Float16*)alloc((size_t)NTOK_PAD * 1024 * 2);
  _Float16* H1 = (_Float16*)alloc((size_t)NTOK_PAD * 256 * 2);
  _Float16* H2 = (_Float16*)alloc((size_t)NTOK_PAD * 64 * 2);
  _Float16* H3 = (_Float16*)alloc((size_t)NTOK_PAD * 16 * 2);
  int* qh = (int*)alloc(NTOK_PAD * 4);
  int* q1 = (int*)alloc(NTOK_PAD * 4);
  int* q2 = (int*)alloc(NTOK_PAD * 4);
  int* q3 = (int*)alloc(NTOK_PAD * 4);
  float* lyH = (float*)alloc(NTOK_PAD * 4);
  float* lyT = (float*)alloc(NTOK_PAD * 4);
  float* pmH = (float*)alloc((size_t)nChH * NTOK_PAD * 4);
  float* psH = (float*)alloc((size_t)nChH * NTOK_PAD * 4);
  float* pm1 = (float*)alloc((size_t)nCh1 * NTOK_PAD * 4);
  float* ps1 = (float*)alloc((size_t)nCh1 * NTOK_PAD * 4);
  float* pm2 = (float*)alloc((size_t)nCh2 * NTOK_PAD * 4);
  float* ps2 = (float*)alloc((size_t)nCh2 * NTOK_PAD * 4);
  float* pm3 = (float*)alloc((size_t)nCh3 * NTOK_PAD * 4);
  float* ps3 = (float*)alloc((size_t)nCh3 * NTOK_PAD * 4);

  setup_q<<<1, NTOK_PAD, 0, stream>>>(lab, qh, q1, q2, q3);

  proj_gemm<<<dim3(32, 12), 256, 0, stream>>>(X, P0, P1, P2, P3, H0, H1, H2, H3);

  auto shBytes = [](int ldsStride) -> size_t {
    return (size_t)64 * ldsStride * 2 + 8 * 64 * 4 * 2 + 64 * 4;
  };

  // head: W = [w0; cluster_w], bias = [b0; cluster_b], 20003 rows
  lse_gemm<<<dim3(8, nChH), 256, shBytes(1032), stream>>>(
      H0, 1024, 1032, W0, B0, HEADV, CW, CB, HEADV + 3, 8, qh, lyH, pmH, psH);
  lse_gemm<<<dim3(8, nCh1), 256, shBytes(264), stream>>>(
      H1, 256, 264, W1, B1, 20000, W1, B1, 20000, 8, q1, lyT, pm1, ps1);
  lse_gemm<<<dim3(8, nCh2), 256, shBytes(72), stream>>>(
      H2, 64, 72, W2, B2, 160000, W2, B2, 160000, 16, q2, lyT, pm2, ps2);
  lse_gemm<<<dim3(8, nCh3), 256, shBytes(24), stream>>>(
      H3, 16, 24, W3, B3, 67735, W3, B3, 67735, 16, q3, lyT, pm3, ps3);

  finalize_nll<<<2, 256, 0, stream>>>(lab, pmH, psH, nChH, pm1, ps1, nCh1, pm2,
                                      ps2, nCh2, pm3, ps3, nCh3, lyH, lyT, out);
}
